// GCNClassifier_31052613550506
// MI455X (gfx1250) — matrix operand order repaired and checked by reading, not compiled, not run on hardware
//
#include <hip/hip_runtime.h>
#include <hip/hip_bf16.h>

// ---------------- problem constants ----------------
constexpr int Bb  = 8;
constexpr int Nn  = 2352;   // 147 * 16
constexpr int Dd  = 512;
constexpr int KpN = 2368;   // N padded to multiple of 32 (K-dim of adj@x)
constexpr int NpM = 2368;   // N padded to multiple of 32 (M-dim for 32-row wave tiles)
constexpr int SL  = 48;
constexpr int Pp  = 512;
constexpr int NL  = 16;
constexpr int RL  = 32;
constexpr int Lw  = 2;

// ---------------- WMMA types / helpers ----------------
typedef __attribute__((ext_vector_type(16))) __bf16 v16bf;
typedef __attribute__((ext_vector_type(8)))  __bf16 v8bf;
typedef __attribute__((ext_vector_type(8)))  float  v8f;

__device__ __forceinline__ v8f wmma_bf16(v16bf a, v16bf b, v8f c) {
  return __builtin_amdgcn_wmma_f32_16x16x32_bf16(
      false, a, false, b, (short)0, c, false, false);
}

// A fragment: 16x32 bf16 tile, row-major, leading dim lda.
// lane<16 -> row=lane, K={0..7,16..23}; lane>=16 -> row=lane-16, K={8..15,24..31}
__device__ __forceinline__ v16bf load_a_frag(const __bf16* Abase, int lda, int kb, int lane) {
  const int m  = lane & 15;
  const int ko = (lane >> 4) << 3;
  const __bf16* p = Abase + (size_t)m * lda + kb + ko;
  v8bf lo = *(const v8bf*)p;
  v8bf hi = *(const v8bf*)(p + 16);
  return __builtin_shufflevector(lo, hi, 0,1,2,3,4,5,6,7,8,9,10,11,12,13,14,15);
}

// B fragment from B^T (N x K row-major, ld ldb).
// lane<16 -> col=lane, K={0..7,16..23} ; lane>=16 -> col=lane-16, K={8..15,24..31}
__device__ __forceinline__ v16bf load_b_frag(const __bf16* BTbase, int ldb, int kb, int lane) {
  const int n  = lane & 15;
  const int kh = (lane >> 4) << 3;
  const __bf16* p = BTbase + (size_t)n * ldb + kb + kh;
  v8bf lo = *(const v8bf*)p;
  v8bf hi = *(const v8bf*)(p + 16);
  return __builtin_shufflevector(lo, hi, 0,1,2,3,4,5,6,7,8,9,10,11,12,13,14,15);
}

// 32x64 C macro-tile per wave: 2 A frags x 4 B frags -> 8 WMMAs per K-step
__device__ __forceinline__ void gemm_32x64(const __bf16* A, int lda,
                                           const __bf16* BT, int ldb,
                                           int K, int lane, v8f acc[2][4]) {
  for (int kb = 0; kb < K; kb += 32) {
    v16bf a0 = load_a_frag(A,                     lda, kb, lane);
    v16bf a1 = load_a_frag(A + (size_t)16 * lda,  lda, kb, lane);
    v16bf b0 = load_b_frag(BT,                    ldb, kb, lane);
    v16bf b1 = load_b_frag(BT + (size_t)16 * ldb, ldb, kb, lane);
    v16bf b2 = load_b_frag(BT + (size_t)32 * ldb, ldb, kb, lane);
    v16bf b3 = load_b_frag(BT + (size_t)48 * ldb, ldb, kb, lane);
    acc[0][0] = wmma_bf16(a0, b0, acc[0][0]);
    acc[0][1] = wmma_bf16(a0, b1, acc[0][1]);
    acc[0][2] = wmma_bf16(a0, b2, acc[0][2]);
    acc[0][3] = wmma_bf16(a0, b3, acc[0][3]);
    acc[1][0] = wmma_bf16(a1, b0, acc[1][0]);
    acc[1][1] = wmma_bf16(a1, b1, acc[1][1]);
    acc[1][2] = wmma_bf16(a1, b2, acc[1][2]);
    acc[1][3] = wmma_bf16(a1, b3, acc[1][3]);
  }
}

// tile bookkeeping: 74 m-tiles (32 rows) x 8 n-quads (64 cols) = 592 waves = 74 blocks x 8
constexpr int MT  = NpM / 32;   // 74
constexpr int NQ  = Dd / 64;    // 8

// ---------------- small prep kernels ----------------
__global__ void k_rowsum(const float* __restrict__ adj, float* __restrict__ denom) {
  __shared__ float red[256];
  const int row = blockIdx.x;                  // b*N + n
  const float* p = adj + (size_t)row * Nn;
  float s = 0.f;
  for (int m = threadIdx.x; m < Nn; m += 256) s += p[m];
  red[threadIdx.x] = s; __syncthreads();
  for (int off = 128; off > 0; off >>= 1) {
    if ((int)threadIdx.x < off) red[threadIdx.x] += red[threadIdx.x + off];
    __syncthreads();
  }
  if (threadIdx.x == 0) denom[row] = red[0] + 1.0f;
}

// adj_bf[b][n][m], n < NpM, m < KpN; zero pad outside [Nn,Nn)
__global__ void k_stage_adj(const float* __restrict__ adj, __bf16* __restrict__ dst) {
  const size_t total = (size_t)Bb * NpM * KpN;
  for (size_t i = (size_t)blockIdx.x * blockDim.x + threadIdx.x; i < total;
       i += (size_t)gridDim.x * blockDim.x) {
    const int m = (int)(i % KpN);
    const size_t bn = i / KpN;
    const int n = (int)(bn % NpM);
    const size_t b = bn / NpM;
    dst[i] = (m < Nn && n < Nn) ? (__bf16)adj[(b * Nn + n) * (size_t)Nn + m] : (__bf16)0.0f;
  }
}

// xT_bf[b][d][n] = x[b][n][d], K padded to KpN with zeros
__global__ void k_stage_xT(const float* __restrict__ x, __bf16* __restrict__ dst) {
  const size_t total = (size_t)Bb * Dd * KpN;
  for (size_t i = (size_t)blockIdx.x * blockDim.x + threadIdx.x; i < total;
       i += (size_t)gridDim.x * blockDim.x) {
    const int n = (int)(i % KpN);
    const size_t bd = i / KpN;
    const int d = (int)(bd % Dd);
    const size_t b = bd / Dd;
    dst[i] = (n < Nn) ? (__bf16)x[(b * Nn + n) * (size_t)Dd + d] : (__bf16)0.0f;
  }
}

// dst (Ncol x K) = transpose(src (K x Ncol)), converted to bf16
__global__ void k_transpose_bf(const float* __restrict__ src, __bf16* __restrict__ dst,
                               int K, int Ncol) {
  const size_t total = (size_t)K * Ncol;
  for (size_t i = (size_t)blockIdx.x * blockDim.x + threadIdx.x; i < total;
       i += (size_t)gridDim.x * blockDim.x) {
    const int k = (int)(i % K);
    const int n = (int)(i / K);
    dst[i] = (__bf16)src[(size_t)k * Ncol + n];
  }
}

// ---------------- GEMM stage 1: Ax = adj @ x ; apx = bf16(Ax + x) ----------------
__global__ __launch_bounds__(256)
void k_gemm_s1(const __bf16* __restrict__ adj_bf,
               const __bf16* __restrict__ xT_bf,
               const float*  __restrict__ x_f32,   // layer0: inputs (else null)
               const __bf16* __restrict__ x_bf,    // layer1: gcat_bf cols 0:512, NpM rows, ld 2D
               __bf16* __restrict__ apx_bf) {
  const int lane = threadIdx.x & 31;
  const int wt = blockIdx.x * 8 + (threadIdx.x >> 5);   // 0..591
  const int mTile = wt >> 3;
  const int nQ = wt & 7;
  const int b = blockIdx.y;

  const __bf16* A  = adj_bf + (size_t)b * NpM * KpN + (size_t)mTile * 32 * KpN;
  const __bf16* BT = xT_bf  + (size_t)b * Dd * KpN + (size_t)nQ * 64 * KpN;
  v8f acc[2][4] = {};
  gemm_32x64(A, KpN, BT, KpN, KpN, lane, acc);

  const int cl = lane & 15;
  const int rb = (lane >> 4) << 3;
  __bf16* O = apx_bf + (size_t)b * NpM * Dd;
  for (int i = 0; i < 2; ++i)
    for (int r = 0; r < 8; ++r) {
      const int row = mTile * 32 + i * 16 + rb + r;
      if (row >= Nn) continue;                      // uniform per 16-row sub-tile
      for (int j = 0; j < 4; ++j) {
        const int col = nQ * 64 + j * 16 + cl;
        float xv;
        if (x_f32) xv = x_f32[((size_t)b * Nn + row) * Dd + col];
        else       xv = (float)x_bf[((size_t)b * NpM + row) * (2 * Dd) + col];
        O[(size_t)row * Dd + col] = (__bf16)(acc[i][j][r] + xv);
      }
    }
}

// ---------------- GEMM stage 2: g = relu(((Ax+x)@W + 2b) / denom) ----------------
__global__ __launch_bounds__(256)
void k_gemm_s2(const __bf16* __restrict__ apx_bf,
               const __bf16* __restrict__ WT,
               const float*  __restrict__ Wb,
               const float*  __restrict__ denom,
               __bf16* __restrict__ gcat_bf,       // [B][NpM][2D], write at layerOff
               __bf16* __restrict__ xT_bf,         // next layer x^T (ld KpN)
               int layerOff, int writeXT) {
  const int lane = threadIdx.x & 31;
  const int wt = blockIdx.x * 8 + (threadIdx.x >> 5);
  const int mTile = wt >> 3;
  const int nQ = wt & 7;
  const int b = blockIdx.y;

  const __bf16* A  = apx_bf + (size_t)b * NpM * Dd + (size_t)mTile * 32 * Dd;
  const __bf16* BT = WT + (size_t)nQ * 64 * Dd;
  v8f acc[2][4] = {};
  gemm_32x64(A, Dd, BT, Dd, Dd, lane, acc);

  const int cl = lane & 15;
  const int rb = (lane >> 4) << 3;
  float bias[4];
  for (int j = 0; j < 4; ++j) bias[j] = 2.0f * Wb[nQ * 64 + j * 16 + cl];
  for (int i = 0; i < 2; ++i)
    for (int r = 0; r < 8; ++r) {
      const int row = mTile * 32 + i * 16 + rb + r;
      if (row >= Nn) continue;
      const float dn = denom[(size_t)b * Nn + row];
      __bf16* g = gcat_bf + ((size_t)b * NpM + row) * (2 * Dd) + layerOff;
      for (int j = 0; j < 4; ++j) {
        const int col = nQ * 64 + j * 16 + cl;
        float v = (acc[i][j][r] + bias[j]) / dn;
        v = fmaxf(v, 0.0f);
        g[col] = (__bf16)v;
        if (writeXT) xT_bf[((size_t)b * Dd + col) * KpN + row] = (__bf16)v;
      }
    }
}

// ---------------- GEMM stage 3: logits = [g0,g1] @ out_w + out_b ----------------
__global__ __launch_bounds__(256)
void k_gemm_s3(const __bf16* __restrict__ gcat_bf,
               const __bf16* __restrict__ outWT,   // (D x 2D) = out_w^T
               const float*  __restrict__ out_b,
               float* __restrict__ logits) {
  const int lane = threadIdx.x & 31;
  const int wt = blockIdx.x * 8 + (threadIdx.x >> 5);
  const int mTile = wt >> 3;
  const int nQ = wt & 7;
  const int b = blockIdx.y;
  constexpr int K = 2 * Dd;

  const __bf16* A  = gcat_bf + (size_t)b * NpM * K + (size_t)mTile * 32 * K;
  const __bf16* BT = outWT + (size_t)nQ * 64 * K;
  v8f acc[2][4] = {};
  gemm_32x64(A, K, BT, K, K, lane, acc);

  const int cl = lane & 15;
  const int rb = (lane >> 4) << 3;
  float bv[4];
  for (int j = 0; j < 4; ++j) bv[j] = out_b[nQ * 64 + j * 16 + cl];
  for (int i = 0; i < 2; ++i)
    for (int r = 0; r < 8; ++r) {
      const int row = mTile * 32 + i * 16 + rb + r;
      if (row >= Nn) continue;
      float* o = logits + ((size_t)b * Nn + row) * Dd;
      for (int j = 0; j < 4; ++j)
        o[nQ * 64 + j * 16 + cl] = acc[i][j][r] + bv[j];
    }
}

// ---------------- NER head ----------------
__global__ void k_ner(const float* __restrict__ ner, const float* __restrict__ logits,
                      const float* __restrict__ nw, const float* __restrict__ nb,
                      float* __restrict__ out) {
  __shared__ float part[256];
  const int bt = blockIdx.x;
  const int b = bt / SL, t = bt - b * SL;
  const int label = threadIdx.x & 15;
  const int chunk = threadIdx.x >> 4;
  const float* nrow = ner + ((size_t)b * SL + t) * Dd;
  const float* lrow = logits + ((size_t)b * Nn + t) * Dd;
  float s = 0.f;
  for (int kk = 0; kk < 64; ++kk) {
    const int k = chunk * 64 + kk;
    const float v = (k < Dd) ? nrow[k] : lrow[k - Dd];
    s += v * nw[(size_t)k * NL + label];
  }
  part[threadIdx.x] = s; __syncthreads();
  if (threadIdx.x < NL) {
    float tot = nb[threadIdx.x];
    for (int c = 0; c < 16; ++c) tot += part[c * 16 + threadIdx.x];
    out[((size_t)b * SL + t) * NL + threadIdx.x] = tot;
  }
}

// ---------------- RE head ----------------
__global__ void k_re(const float* __restrict__ re_feat, const float* __restrict__ logits,
                     const float* __restrict__ rw, const float* __restrict__ rb,
                     const int* __restrict__ pair_sent, const int* __restrict__ head_start,
                     const int* __restrict__ tail_start, float* __restrict__ out) {
  __shared__ float avg[Dd];
  __shared__ float part[256];
  const int p = blockIdx.x;
  const int b  = pair_sent[p];
  const int hs = head_start[p];
  const int ts = tail_start[p];
  for (int d = threadIdx.x; d < Dd; d += 256) {
    float s = 0.f;
    for (int l1 = 0; l1 < Lw; ++l1)
      for (int l2 = 0; l2 < Lw; ++l2) {
        const int r = (hs + l1 + 1) * SL + (ts + l2);
        s += logits[((size_t)b * Nn + r) * Dd + d];
      }
    avg[d] = s * (1.0f / (Lw * Lw));
  }
  __syncthreads();
  const int label = threadIdx.x & 31;
  const int chunk = threadIdx.x >> 5;
  float s = 0.f;
  for (int kk = 0; kk < 128; ++kk) {
    const int k = chunk * 128 + kk;
    const float v = (k < Dd) ? re_feat[(size_t)p * Dd + k] : avg[k - Dd];
    s += v * rw[(size_t)k * RL + label];
  }
  part[threadIdx.x] = s; __syncthreads();
  if (threadIdx.x < RL) {
    float tot = rb[threadIdx.x];
    for (int c = 0; c < 8; ++c) tot += part[c * 32 + threadIdx.x];
    out[(size_t)p * RL + threadIdx.x] = tot;
  }
}

// ---------------- host launch ----------------
extern "C" void kernel_launch(void* const* d_in, const int* in_sizes, int n_in,
                              void* d_out, int out_size, void* d_ws, size_t ws_size,
                              hipStream_t stream) {
  const float* inputs  = (const float*)d_in[0];
  const float* ner     = (const float*)d_in[1];
  const float* re_feat = (const float*)d_in[2];
  const float* adj     = (const float*)d_in[3];
  const float* W0_w    = (const float*)d_in[4];
  const float* W0_b    = (const float*)d_in[5];
  const float* W1_w    = (const float*)d_in[6];
  const float* W1_b    = (const float*)d_in[7];
  const float* out_w   = (const float*)d_in[8];
  const float* out_b   = (const float*)d_in[9];
  const float* ner_w   = (const float*)d_in[10];
  const float* ner_b   = (const float*)d_in[11];
  const float* re_w    = (const float*)d_in[12];
  const float* re_b    = (const float*)d_in[13];
  const int* pair_sent  = (const int*)d_in[14];
  const int* head_start = (const int*)d_in[15];
  const int* tail_start = (const int*)d_in[16];
  (void)in_sizes; (void)n_in; (void)out_size; (void)ws_size;

  char* ws = (char*)d_ws;
  size_t off = 0;
  auto carve = [&](size_t bytes) {
    void* p = ws + off;
    off += (bytes + 255) & ~(size_t)255;
    return p;
  };
  float*  denom   = (float*) carve((size_t)Bb * Nn * 4);
  __bf16* adj_bf  = (__bf16*)carve((size_t)Bb * NpM * KpN * 2);
  __bf16* xT_bf   = (__bf16*)carve((size_t)Bb * Dd * KpN * 2);
  __bf16* apx_bf  = (__bf16*)carve((size_t)Bb * NpM * Dd * 2);
  __bf16* W0T     = (__bf16*)carve((size_t)Dd * Dd * 2);
  __bf16* W1T     = (__bf16*)carve((size_t)Dd * Dd * 2);
  __bf16* outWT   = (__bf16*)carve((size_t)Dd * 2 * Dd * 2);
  __bf16* gcat_bf = (__bf16*)carve((size_t)Bb * NpM * 2 * Dd * 2);
  float*  logits  = (float*) carve((size_t)Bb * Nn * Dd * 4);

  float* out_ner = (float*)d_out;
  float* out_re  = out_ner + (size_t)Bb * SL * NL;

  // prep
  k_rowsum      <<<Bb * Nn, 256, 0, stream>>>(adj, denom);
  k_stage_adj   <<<8192, 256, 0, stream>>>(adj, adj_bf);
  k_stage_xT    <<<2048, 256, 0, stream>>>(inputs, xT_bf);
  k_transpose_bf<<<512,  256, 0, stream>>>(W0_w, W0T, Dd, Dd);
  k_transpose_bf<<<512,  256, 0, stream>>>(W1_w, W1T, Dd, Dd);
  k_transpose_bf<<<1024, 256, 0, stream>>>(out_w, outWT, 2 * Dd, Dd);

  const dim3 ggrid(MT, Bb);                 // 74 blocks x 8 waves = 592 wave-tiles exactly
  // layer 0
  k_gemm_s1<<<ggrid, 256, 0, stream>>>(adj_bf, xT_bf, inputs, nullptr, apx_bf);
  k_gemm_s2<<<ggrid, 256, 0, stream>>>(apx_bf, W0T, W0_b, denom, gcat_bf, xT_bf, 0, 1);
  // layer 1 (x = g0, read from gcat_bf cols 0:512)
  k_gemm_s1<<<ggrid, 256, 0, stream>>>(adj_bf, xT_bf, nullptr, gcat_bf, apx_bf);
  k_gemm_s2<<<ggrid, 256, 0, stream>>>(apx_bf, W1T, W1_b, denom, gcat_bf, xT_bf, Dd, 0);
  // output projection
  k_gemm_s3<<<ggrid, 256, 0, stream>>>(gcat_bf, outWT, out_b, logits);
  // heads
  k_ner<<<Bb * SL, 256, 0, stream>>>(ner, logits, ner_w, ner_b, out_ner);
  k_re <<<Pp,      256, 0, stream>>>(re_feat, logits, re_w, re_b,
                                     pair_sent, head_start, tail_start, out_re);
}
